// LSTMModule_7344394076374
// MI455X (gfx1250) — matrix operand order repaired and checked by reading, not compiled, not run on hardware
//
#include <hip/hip_runtime.h>
#include <hip/hip_bf16.h>
#include <stdint.h>

// Problem dims (from reference)
#define Bx 8192
#define Tx 365
#define Ix 3
#define Hx 34
#define Gx 136          // 4*H
#define KP 64           // K (hidden) padded to 2 x 32 for v_wmma_f32_16x16x32_bf16
#define NP 160          // gate dim padded to 10 N-tiles of 16 (tile 9 is all-zero pad)
#define GST 140         // gates LDS row stride (floats): 8*140 % 64 == 32 -> no bank clash row0 vs row8
#define HST 48          // h LDS row stride in bf16 entries (96B); covers K 0..47, 16B-aligned subloads
#define CST 35          // c LDS row stride (floats)
#define TILES_PER_BLOCK 2
#define ROWS_PER_BLOCK 32   // 2 tiles x 16 batch rows
#define THREADS 128         // 4 waves: wave pair (2 waves) cooperates on one 16-row tile

typedef __attribute__((ext_vector_type(16))) __bf16 v16bf;
typedef __attribute__((ext_vector_type(8)))  float  v8f;

union Frag {                 // 16 bf16 = 32B per lane = one WMMA A/B operand
    v16bf        v;
    uint4        q[2];
    unsigned int u[8];
};

__device__ __forceinline__ unsigned short f2bf(float f) {
    unsigned int u = __builtin_bit_cast(unsigned int, f);
    return (unsigned short)((u + 0x7FFFu + ((u >> 16) & 1u)) >> 16); // RNE
}
__device__ __forceinline__ float fsig(float x) {
    return __builtin_amdgcn_rcpf(1.0f + __expf(-x));
}
__device__ __forceinline__ float ftanh(float x) {
    return 1.0f - 2.0f * __builtin_amdgcn_rcpf(__expf(2.0f * x) + 1.0f);
}

// x [B,T,I] -> xT [T,B,I] so the per-timestep x tile is a coalesced block read.
__global__ void xT_kernel(const float* __restrict__ x, float* __restrict__ xT) {
    int64_t idx = (int64_t)blockIdx.x * blockDim.x + threadIdx.x;
    const int64_t total = (int64_t)Bx * Tx * Ix;
    if (idx < total) {
        int     c  = (int)(idx % Ix);
        int64_t bt = idx / Ix;
        int     b  = (int)(bt % Bx);
        int     t  = (int)(bt / Bx);
        xT[idx] = x[((int64_t)b * Tx + t) * Ix + c];
    }
}

__global__ __launch_bounds__(THREADS, 1) void lstm_fused(
    const float* __restrict__ x,      // [B,T,I]
    const float* __restrict__ xT,     // [T,B,I] (valid iff useXT)
    int useXT,
    const float* __restrict__ fc0_w,  // [H,I]
    const float* __restrict__ fc0_b,  // [H]
    const float* __restrict__ w_ih,   // [4H,H]
    const float* __restrict__ w_hh,   // [4H,H]
    const float* __restrict__ b_ih,   // [4H]
    const float* __restrict__ b_hh,   // [4H]
    float* __restrict__ out)          // [B,T,H]
{
    // LDS: ~47 KB/block -> multiple blocks per WGP (320 KB)
    __shared__ __align__(16) unsigned short s_w[NP * KP];                    // w_hh as bf16, [gate][k], zero padded
    __shared__ float s_wc[4 * Gx];                                           // combined input map + folded bias
    __shared__ __align__(16) unsigned short s_h[TILES_PER_BLOCK * 16 * HST]; // h state, bf16, A-fragment layout friendly
    __shared__ float s_c[TILES_PER_BLOCK * 16 * CST];                        // c state f32
    __shared__ float s_g[TILES_PER_BLOCK * 16 * GST];                        // per-step gates (WMMA D scatter target)
    __shared__ float s_x[ROWS_PER_BLOCK * Ix];                               // x_t tile

    const int tid      = threadIdx.x;
    const int lane     = tid & 31;
    const int wid      = tid >> 5;     // 0..3
    const int tile     = wid >> 1;     // 0..1 : which 16-row batch tile
    const int half     = wid & 1;      // 0 -> N tiles 0..4, 1 -> N tiles 5..9
    const int rowBase  = blockIdx.x * ROWS_PER_BLOCK;
    const int tileRow0 = tile * 16;

    // ---- one-time staging -------------------------------------------------
    // B matrix: s_w[n][k] = w_hh[n][k] (bf16), zeros for k>=34 or n>=136
    for (int idx = tid; idx < NP * KP; idx += THREADS) {
        int g = idx / KP, k = idx - g * KP;
        s_w[idx] = (g < Gx && k < Hx) ? f2bf(w_hh[g * Hx + k]) : (unsigned short)0;
    }
    // Combined input-side map: gates_x[b,g] = sum_i x_i * Wc[i][g] + Wc[3][g]
    for (int g = tid; g < Gx; g += THREADS) {
        float a0 = 0.f, a1 = 0.f, a2 = 0.f, a3 = 0.f;
        for (int h = 0; h < Hx; ++h) {
            float w = w_ih[g * Hx + h];
            a0 += w * fc0_w[h * Ix + 0];
            a1 += w * fc0_w[h * Ix + 1];
            a2 += w * fc0_w[h * Ix + 2];
            a3 += w * fc0_b[h];
        }
        s_wc[0 * Gx + g] = a0;
        s_wc[1 * Gx + g] = a1;
        s_wc[2 * Gx + g] = a2;
        s_wc[3 * Gx + g] = a3 + b_ih[g] + b_hh[g];
    }
    // h0 = c0 = 0 (pad entries too)
    for (int i = tid; i < TILES_PER_BLOCK * 16 * HST; i += THREADS) s_h[i] = 0;
    for (int i = tid; i < TILES_PER_BLOCK * 16 * CST; i += THREADS) s_c[i] = 0.f;
    __syncthreads();

    // ---- register-resident B fragments (w_hh), 5 N-tiles x 2 K-halves -----
    // B (32x16): lanes 0-15 hold K k*32+[0..15] of column N=n*16+lane,
    //            lanes 16-31 hold K k*32+[16..31] of column N=n*16+lane-16.
    Frag bf[5][2];
#pragma unroll
    for (int j = 0; j < 5; ++j) {
        const int n = half * 5 + j;
        const unsigned short* wrow = s_w + (n * 16 + (lane & 15)) * KP + ((lane & 16) ? 16 : 0);
#pragma unroll
        for (int k = 0; k < 2; ++k) {
            bf[j][k].q[0] = *(const uint4*)(wrow + k * 32);
            bf[j][k].q[1] = *(const uint4*)(wrow + k * 32 + 8);
        }
    }

    float* gt = s_g + tileRow0 * GST;          // this tile's gate scratch
    unsigned short* hw = s_h + tileRow0 * HST; // this tile's h rows

    // ---- sequential recurrence -------------------------------------------
    for (int t = 0; t < Tx; ++t) {
        __syncthreads(); // h/c from prev step visible across wave pair; s_x safe to refill
        if (useXT) {
            const float* src = xT + (int64_t)t * Bx * Ix + (int64_t)rowBase * Ix;
            for (int i = tid; i < ROWS_PER_BLOCK * Ix; i += THREADS) s_x[i] = src[i];
        } else {
            for (int i = tid; i < ROWS_PER_BLOCK * Ix; i += THREADS) {
                int r = i / Ix, c2 = i - r * Ix;
                s_x[i] = x[((int64_t)(rowBase + r) * Tx + t) * Ix + c2];
            }
        }
        __syncthreads();

        // A fragments from bf16 h. K order used here is contiguous (same as B fragments):
        // lanes 0-15: elems = K{0..15}; lanes 16-31: K{16..31}; M = lane%16.
        const unsigned short* hrow = s_h + (tileRow0 + (lane & 15)) * HST;
        const int o = (lane & 16) ? 16 : 0;
        Frag a0, a1;
        a0.q[0] = *(const uint4*)(hrow + o);
        a0.q[1] = *(const uint4*)(hrow + o + 8);
#pragma unroll
        for (int e = 0; e < 8; ++e) a1.u[e] = 0;
        // second K-half: only K=32,33 (h[32],h[33]) are nonzero -> elems 0,1 of lanes 0-15
        a1.u[0] = (lane & 16) ? 0u : *(const unsigned int*)(hrow + 32);

        // gates = h @ w_hh^T, 16x16 tile per n, K=64 in two bf16 WMMAs
#pragma unroll
        for (int j = 0; j < 5; ++j) {
            const int n = half * 5 + j;
            v8f acc = {};
            acc = __builtin_amdgcn_wmma_f32_16x16x32_bf16(false, a0.v, false, bf[j][0].v,
                                                          (short)0, acc, false, false);
            acc = __builtin_amdgcn_wmma_f32_16x16x32_bf16(false, a1.v, false, bf[j][1].v,
                                                          (short)0, acc, false, false);
            if (n < 9) { // tile 9 is pad
                const int col = n * 16 + (lane & 15);
                const int r0  = (lane & 16) ? 8 : 0; // C/D layout: lanes 16-31 hold M=8..15
#pragma unroll
                for (int r = 0; r < 8; ++r) gt[(r0 + r) * GST + col] = acc[r];
            }
        }
        __syncthreads();

        // activations + state update; wave pair splits the 16*34 (row,hid) pairs
        for (int p = lane + 32 * half; p < 16 * Hx; p += 64) {
            const int row = p / Hx;
            const int hid = p - row * Hx;
            const int brow = tileRow0 + row;
            const float x0 = s_x[brow * 3 + 0];
            const float x1 = s_x[brow * 3 + 1];
            const float x2 = s_x[brow * 3 + 2];
            const int gI = hid, gF = hid + Hx, gG = hid + 2 * Hx, gO = hid + 3 * Hx;
            const float* grow = gt + row * GST;
            float vi = grow[gI] + s_wc[gI] * x0 + s_wc[Gx + gI] * x1 + s_wc[2 * Gx + gI] * x2 + s_wc[3 * Gx + gI];
            float vf = grow[gF] + s_wc[gF] * x0 + s_wc[Gx + gF] * x1 + s_wc[2 * Gx + gF] * x2 + s_wc[3 * Gx + gF];
            float vg = grow[gG] + s_wc[gG] * x0 + s_wc[Gx + gG] * x1 + s_wc[2 * Gx + gG] * x2 + s_wc[3 * Gx + gG];
            float vo = grow[gO] + s_wc[gO] * x0 + s_wc[Gx + gO] * x1 + s_wc[2 * Gx + gO] * x2 + s_wc[3 * Gx + gO];
            const float ig = fsig(vi), fg = fsig(vf), gg = ftanh(vg), og = fsig(vo);
            const int ci = brow * CST + hid;
            const float cNew = fg * s_c[ci] + ig * gg;
            const float hNew = og * ftanh(cNew);
            s_c[ci] = cNew;
            hw[row * HST + hid] = f2bf(hNew);
            out[(int64_t)(rowBase + brow) * Tx * Hx + (int64_t)t * Hx + hid] = hNew;
        }
    }
}

extern "C" void kernel_launch(void* const* d_in, const int* in_sizes, int n_in,
                              void* d_out, int out_size, void* d_ws, size_t ws_size,
                              hipStream_t stream) {
    const float* x     = (const float*)d_in[0];
    const float* fc0_w = (const float*)d_in[1];
    const float* fc0_b = (const float*)d_in[2];
    const float* w_ih  = (const float*)d_in[3];
    const float* w_hh  = (const float*)d_in[4];
    const float* b_ih  = (const float*)d_in[5];
    const float* b_hh  = (const float*)d_in[6];
    float* out = (float*)d_out;

    const size_t xTbytes = (size_t)Bx * Tx * Ix * sizeof(float); // ~36 MB
    const int useXT = (ws_size >= xTbytes) ? 1 : 0;
    float* xT = (float*)d_ws;
    if (useXT) {
        const int64_t total = (int64_t)Bx * Tx * Ix;
        const int thr = 256;
        const int blocks = (int)((total + thr - 1) / thr);
        xT_kernel<<<blocks, thr, 0, stream>>>(x, xT);
    }
    lstm_fused<<<Bx / ROWS_PER_BLOCK, THREADS, 0, stream>>>(
        x, xT, useXT, fc0_w, fc0_b, w_ih, w_hh, b_ih, b_hh, out);
}
